// EncoderPositional_AIAYN_31129922961551
// MI455X (gfx1250) — compile-verified, matrix operand order checked by reading, not run on hardware
//
#include <hip/hip_runtime.h>
#include <hip/hip_bf16.h>

// ---------------------------------------------------------------------------
// Types for CDNA5 WMMA
// ---------------------------------------------------------------------------
typedef __attribute__((ext_vector_type(16))) __bf16        v16bf;
typedef __attribute__((ext_vector_type(8)))  float         v8f;
typedef __attribute__((ext_vector_type(4)))  float         v4f;
typedef __attribute__((ext_vector_type(4)))  unsigned int  v4u;
typedef __attribute__((ext_vector_type(8)))  unsigned short v8us;

union FragBF16 {
    v16bf v;
    struct { v4u lo, hi; } u;
};

__device__ __forceinline__ unsigned short f2bf(float f) {
    unsigned int u = __float_as_uint(f);
    // round-to-nearest-even truncation to bf16
    unsigned int r = u + 0x7FFFu + ((u >> 16) & 1u);
    return (unsigned short)(r >> 16);
}

// ---------------------------------------------------------------------------
// Sizes (compile-time; match the reference exactly)
// ---------------------------------------------------------------------------
#define SEQ     4096
#define WORD_D  512
#define HIDDEN  1024   // WORD_D + POS_D

// GEMM tiling
#define BLK_M   128
#define BLK_N   128
#define BLK_K   32
#define LDS_STRIDE 40          // padded row stride in bf16 elems (80 B) - bank spread
#define TILE_ELEMS (128 * LDS_STRIDE)   // one 128x32 tile (padded) = 5120 elems

// ---------------------------------------------------------------------------
// Prepass 1: W [1024,1024] f32 -> bf16
// ---------------------------------------------------------------------------
__global__ void convert_W_kernel(const float* __restrict__ W,
                                 __bf16* __restrict__ Wbf) {
    size_t i = ((size_t)blockIdx.x * blockDim.x + threadIdx.x) * 8;
    v4f x0 = *(const v4f*)(W + i);
    v4f x1 = *(const v4f*)(W + i + 4);
    v8us o;
    o[0] = f2bf(x0[0]); o[1] = f2bf(x0[1]); o[2] = f2bf(x0[2]); o[3] = f2bf(x0[3]);
    o[4] = f2bf(x1[0]); o[5] = f2bf(x1[1]); o[6] = f2bf(x1[2]); o[7] = f2bf(x1[3]);
    *(v8us*)(Wbf + i) = o;
}

// ---------------------------------------------------------------------------
// Prepass 2: c[m, 0:512] = word_emb[inputs[m]], c[m, 512:1024] = pos_emb[m]
// ---------------------------------------------------------------------------
__global__ void build_c_kernel(const int* __restrict__ inputs,
                               const float* __restrict__ word_emb,
                               const float* __restrict__ pos_emb,
                               __bf16* __restrict__ cbf) {
    int m   = blockIdx.x;          // 0..4095
    int col = threadIdx.x * 8;     // 0..1016
    const float* src;
    if (col < WORD_D) {
        src = word_emb + (size_t)inputs[m] * WORD_D + col;
    } else {
        src = pos_emb + (size_t)m * WORD_D + (col - WORD_D);
    }
    v4f x0 = *(const v4f*)src;
    v4f x1 = *(const v4f*)(src + 4);
    v8us o;
    o[0] = f2bf(x0[0]); o[1] = f2bf(x0[1]); o[2] = f2bf(x0[2]); o[3] = f2bf(x0[3]);
    o[4] = f2bf(x1[0]); o[5] = f2bf(x1[1]); o[6] = f2bf(x1[2]); o[7] = f2bf(x1[3]);
    *(v8us*)(cbf + (size_t)m * HIDDEN + col) = o;
}

// ---------------------------------------------------------------------------
// GEMM: out[m,n] = sigmoid( sum_k c[m,k] * W[n,k] + b[n] )
//
// Block = 256 threads (8 waves).  Block tile = 128(M) x 128(N), K step 32.
// Wave grid 4(M) x 2(N): each wave computes 32x64 = 2x4 WMMA 16x16 tiles
// -> 8 v_wmma per K-step per wave.
//
// LDS: double-buffered A(128x32) + B(128x32) bf16 tiles, rows padded to
// 40 elems (80 B) so the 16-lane b128 fragment reads hit distinct banks.
// Pipeline: global loads for step s+1 issue before compute of step s.
//
// Fragment layout (16-bit A/B, ISA 7.12.2): lane = h*16 + r,
//   lo 8 elems at K = h*8, hi 8 elems at K = 16 + h*8, row = r.
// C/D layout: VGPR v, lanes 0-15 -> M=v, lanes 16-31 -> M=v+8; N = lane&15.
// ---------------------------------------------------------------------------
__global__ void
__launch_bounds__(256)
gemm_bias_sigmoid_kernel(const __bf16* __restrict__ A,    // c_bf [4096,1024]
                         const __bf16* __restrict__ B,    // W_bf [1024,1024]
                         const float*  __restrict__ bias, // [1024]
                         float* __restrict__ out) {       // [4096,1024]
    __shared__ __bf16 smem[2 * 2 * TILE_ELEMS];  // 40 KB (of 320 KB/WGP)

    const int tid  = threadIdx.x;
    const int lane = tid & 31;
    const int wave = tid >> 5;
    const int wm   = wave & 3;   // 0..3 -> 32-row M slice
    const int wn   = wave >> 2;  // 0..1 -> 64-col N slice

    const int m0 = blockIdx.y * BLK_M;
    const int n0 = blockIdx.x * BLK_N;

    // cooperative global->LDS staging assignment: 2 tiles * 8 KB, 32 B/thread/tile
    const int lrow = tid & 127;
    const int lseg = tid >> 7;                       // 0/1 -> 16-elem chunk
    const __bf16* gA = A + (size_t)(m0 + lrow) * HIDDEN + lseg * 16;
    const __bf16* gB = B + (size_t)(n0 + lrow) * HIDDEN + lseg * 16;
    const int soff   = lrow * LDS_STRIDE + lseg * 16;

    // fragment read coordinates
    const int r  = lane & 15;
    const int h8 = (lane >> 4) * 8;
    const int aro = (wm * 32 + r) * LDS_STRIDE;      // + mi*16*LDS_STRIDE
    const int bro = (wn * 64 + r) * LDS_STRIDE;      // + nj*16*LDS_STRIDE

    v8f acc[2][4] = {};
    v4u ra0, ra1, rb0, rb1;

    auto loadG = [&](int k) {
        ra0 = *(const v4u*)(gA + k);
        ra1 = *(const v4u*)(gA + k + 8);
        rb0 = *(const v4u*)(gB + k);
        rb1 = *(const v4u*)(gB + k + 8);
    };
    auto storeS = [&](int buf) {
        __bf16* base = smem + buf * (2 * TILE_ELEMS);
        *(v4u*)(base + soff)                  = ra0;
        *(v4u*)(base + soff + 8)              = ra1;
        *(v4u*)(base + TILE_ELEMS + soff)     = rb0;
        *(v4u*)(base + TILE_ELEMS + soff + 8) = rb1;
    };
    auto compute = [&](int buf) {
        const __bf16* sa = smem + buf * (2 * TILE_ELEMS);
        const __bf16* sb = sa + TILE_ELEMS;
        FragBF16 fa[2], fb[4];
        #pragma unroll
        for (int mi = 0; mi < 2; ++mi) {
            int ro = aro + mi * 16 * LDS_STRIDE;
            fa[mi].u.lo = *(const v4u*)(sa + ro + h8);
            fa[mi].u.hi = *(const v4u*)(sa + ro + 16 + h8);
        }
        #pragma unroll
        for (int nj = 0; nj < 4; ++nj) {
            int ro = bro + nj * 16 * LDS_STRIDE;
            fb[nj].u.lo = *(const v4u*)(sb + ro + h8);
            fb[nj].u.hi = *(const v4u*)(sb + ro + 16 + h8);
        }
        #pragma unroll
        for (int mi = 0; mi < 2; ++mi)
            #pragma unroll
            for (int nj = 0; nj < 4; ++nj)
                acc[mi][nj] = __builtin_amdgcn_wmma_f32_16x16x32_bf16(
                    false, fa[mi].v, false, fb[nj].v, (short)0,
                    acc[mi][nj], false, false);
    };

    const int S = HIDDEN / BLK_K;   // 32 K-steps

    loadG(0);
    storeS(0);
    __syncthreads();

    for (int s = 0; s < S; ++s) {
        if (s + 1 < S) loadG((s + 1) * BLK_K);   // overlap with compute below
        compute(s & 1);
        if (s + 1 < S) {
            __syncthreads();                      // readers of next buf are done
            storeS((s + 1) & 1);
            __syncthreads();                      // stores visible before reads
        }
    }

    // epilogue: bias + sigmoid + store
    #pragma unroll
    for (int mi = 0; mi < 2; ++mi) {
        const int mrow = m0 + wm * 32 + mi * 16 + ((lane >> 4) * 8);
        #pragma unroll
        for (int nj = 0; nj < 4; ++nj) {
            const int n = n0 + wn * 64 + nj * 16 + (lane & 15);
            const float bv = bias[n];
            #pragma unroll
            for (int v = 0; v < 8; ++v) {
                float x = acc[mi][nj][v] + bv;
                out[(size_t)(mrow + v) * HIDDEN + n] =
                    1.0f / (1.0f + __expf(-x));
            }
        }
    }
}

// ---------------------------------------------------------------------------
// Column-mean reduction (two stages)
// ---------------------------------------------------------------------------
__global__ void col_partial_kernel(const float* __restrict__ out,
                                   float* __restrict__ part) {
    int n  = blockIdx.x * 256 + threadIdx.x;  // 0..1023
    int mb = blockIdx.y;                      // 0..7, 512 rows each
    const float* p = out + (size_t)mb * 512 * HIDDEN + n;
    float s = 0.0f;
    #pragma unroll 8
    for (int m = 0; m < 512; ++m) s += p[(size_t)m * HIDDEN];
    part[mb * HIDDEN + n] = s;
}

__global__ void col_final_kernel(const float* __restrict__ part,
                                 float* __restrict__ hid) {
    int n = blockIdx.x * 256 + threadIdx.x;
    float s = 0.0f;
    #pragma unroll
    for (int i = 0; i < 8; ++i) s += part[i * HIDDEN + n];
    hid[n] = s * (1.0f / 4096.0f);
}

// ---------------------------------------------------------------------------
// Launch
// ---------------------------------------------------------------------------
extern "C" void kernel_launch(void* const* d_in, const int* in_sizes, int n_in,
                              void* d_out, int out_size, void* d_ws, size_t ws_size,
                              hipStream_t stream) {
    const int*   inputs   = (const int*)  d_in[0];  // [4096]
    const float* word_emb = (const float*)d_in[1];  // [50257, 512]
    const float* pos_emb  = (const float*)d_in[2];  // [4096, 512]
    const float* W        = (const float*)d_in[3];  // [1024, 1024]
    const float* b        = (const float*)d_in[4];  // [1024]

    float* out = (float*)d_out;                         // [4096*1024]
    float* hid = (float*)d_out + (size_t)SEQ * HIDDEN;  // [1024]

    // Workspace layout
    __bf16* cbf  = (__bf16*)d_ws;                                        // 8 MB
    __bf16* wbf  = (__bf16*)((char*)d_ws + (size_t)SEQ * HIDDEN * 2);    // 2 MB
    float*  part = (float*) ((char*)d_ws + (size_t)SEQ * HIDDEN * 2
                                         + (size_t)HIDDEN * HIDDEN * 2); // 32 KB

    // 1) W f32 -> bf16 : 1024*1024 / (256*8) = 512 blocks
    convert_W_kernel<<<512, 256, 0, stream>>>(W, wbf);

    // 2) gather + concat + convert : one block per row
    build_c_kernel<<<SEQ, 128, 0, stream>>>(inputs, word_emb, pos_emb, cbf);

    // 3) WMMA GEMM + bias + sigmoid : block tile 128x128
    dim3 ggrid(HIDDEN / BLK_N, SEQ / BLK_M);   // (8, 32)
    gemm_bias_sigmoid_kernel<<<ggrid, 256, 0, stream>>>(cbf, wbf, b, out);

    // 4) column mean
    dim3 pgrid(HIDDEN / 256, 8);
    col_partial_kernel<<<pgrid, 256, 0, stream>>>(out, part);
    col_final_kernel<<<HIDDEN / 256, 256, 0, stream>>>(part, hid);
}
